// MHDLoss_31190052503846
// MI455X (gfx1250) — compile-verified, matrix operand order checked by reading, not run on hardware
//
#include <hip/hip_runtime.h>
#include <hip/hip_bf16.h>

typedef __attribute__((ext_vector_type(16))) _Float16 v16h;
typedef __attribute__((ext_vector_type(8)))  _Float16 v8h;
typedef __attribute__((ext_vector_type(8)))  float    v8f;
typedef __attribute__((ext_vector_type(4)))  int      v4i;

#define AS1 __attribute__((address_space(1)))
#define AS3 __attribute__((address_space(3)))

constexpr int NPTS = 131072;
constexpr int HH   = 256;
constexpr int ROWS = 64;            // rows per block (4 waves, 16 rows/wave -> 1 wave/SIMD)
constexpr int NTHR = 128;
constexpr int NBLK = NPTS / ROWS;   // 2048
constexpr float GAMMA = 5.0f / 3.0f;

// LDS layout (in halfs):
//   Wt   : 65536   (256x256 f16, N-major)            128 KB
//   hA   : 16384   (64x256 f16)                      32 KB
//   hB   : 16384                                     32 KB
//   dA[3]: 3x16384 (tangents, updated in place)      96 KB
// then float area: w4s[1536], b4s[6], red[7]         ~6 KB      => ~301 KB total
constexpr int HALFS_TOTAL = 65536 + 2 * 16384 + 3 * 16384;   // 147456 halfs
constexpr int FLOATS_TOTAL = 1536 + 6 + 7;
constexpr size_t SMEM_BYTES = (size_t)HALFS_TOTAL * 2 + (size_t)FLOATS_TOTAL * 4 + 16;

__device__ __forceinline__ float my_tanhf(float x) {
#if defined(__has_builtin)
# if __has_builtin(__builtin_amdgcn_tanhf)
    return __builtin_amdgcn_tanhf(x);
# elif __has_builtin(__builtin_amdgcn_tanh_f32)
    return __builtin_amdgcn_tanh_f32(x);
# else
    return tanhf(x);
# endif
#else
    return tanhf(x);
#endif
}

// Async global->LDS 16-byte copy (CDNA5 GLOBAL_LOAD_ASYNC_TO_LDS_B128, ASYNCcnt).
__device__ __forceinline__ void async_copy_b128(const void* gsrc, void* ldst) {
#if defined(__has_builtin) && __has_builtin(__builtin_amdgcn_global_load_async_to_lds_b128)
    __builtin_amdgcn_global_load_async_to_lds_b128(
        (AS1 v4i*)gsrc, (AS3 v4i*)ldst, 0, 0);
#else
    unsigned loff = (unsigned)(uintptr_t)(AS3 char*)ldst;
    asm volatile("global_load_async_to_lds_b128 %0, %1, off"
                 :: "v"(loff), "v"(gsrc) : "memory");
#endif
}

__device__ __forceinline__ void wait_async0() {
#if defined(__has_builtin) && __has_builtin(__builtin_amdgcn_s_wait_asynccnt)
    __builtin_amdgcn_s_wait_asynccnt(0);
#else
    asm volatile("s_wait_asynccnt 0x0" ::: "memory");
#endif
}

// Load a 16x32 f16 WMMA A/B operand tile from a row/N-major LDS buffer.
// Per ISA layout: lane holds row (lane&15); half j -> K = kc + j + 8*(j>=8) + 8*(lane>=16).
__device__ __forceinline__ v16h lds_tile16(const _Float16* rowBase, int kc, int lane) {
    const int r   = lane & 15;
    const int off = (lane >> 4) << 3;  // 0 or 8
    const _Float16* p = rowBase + r * HH + kc + off;
    v8h lo = *(const v8h*)(p);
    v8h hi = *(const v8h*)(p + 16);
    v16h out;
#pragma unroll
    for (int i = 0; i < 8; ++i) { out[i] = lo[i]; out[8 + i] = hi[i]; }
    return out;
}

// Forward stream: Out(16x256) = tanh(Ain(16x256) @ W + bias), row-major LDS buffers.
__device__ __forceinline__ void forward_stream(
    const _Float16* __restrict__ Ain, const _Float16* __restrict__ Wt,
    _Float16* __restrict__ Out, const float* __restrict__ biasG,
    int lane, int waveRow)
{
    const int nlo = lane & 15;
    const int rhi = (lane >> 4) << 3;
#pragma unroll 1
    for (int j = 0; j < 16; ++j) {
        float bv = biasG[j * 16 + nlo];
        v8f acc;
#pragma unroll
        for (int r = 0; r < 8; ++r) acc[r] = bv;
#pragma unroll
        for (int kc = 0; kc < HH; kc += 32) {
            v16h a = lds_tile16(Ain + waveRow * HH, kc, lane);
            v16h b = lds_tile16(Wt + (j * 16) * HH, kc, lane);
            acc = __builtin_amdgcn_wmma_f32_16x16x32_f16(
                false, a, false, b, (short)0, acc, false, false);
        }
        const int n = j * 16 + nlo;
#pragma unroll
        for (int r = 0; r < 8; ++r) {       // D layout: row = r + 8*(lane>=16)
            const int row = waveRow + rhi + r;
            Out[row * HH + n] = (_Float16)my_tanhf(acc[r]);
        }
    }
}

// Tangent stream, IN PLACE: dBuf <- (1 - Hmask^2) * (dBuf @ W).
// All 16 output tiles are held in registers (j fully unrolled) and written back
// only after every K-read of dBuf has completed -> no ping-pong buffer needed.
__device__ __forceinline__ void tangent_stream_inplace(
    _Float16* __restrict__ dBuf, const _Float16* __restrict__ Wt,
    const _Float16* __restrict__ Hmask, int lane, int waveRow)
{
    const int nlo = lane & 15;
    const int rhi = (lane >> 4) << 3;
    v8h outT[16];
#pragma unroll
    for (int j = 0; j < 16; ++j) {
        v8f acc = {};
#pragma unroll
        for (int kc = 0; kc < HH; kc += 32) {
            v16h a = lds_tile16(dBuf + waveRow * HH, kc, lane);
            v16h b = lds_tile16(Wt + (j * 16) * HH, kc, lane);
            acc = __builtin_amdgcn_wmma_f32_16x16x32_f16(
                false, a, false, b, (short)0, acc, false, false);
        }
        const int n = j * 16 + nlo;
#pragma unroll
        for (int r = 0; r < 8; ++r) {
            const int row = waveRow + rhi + r;
            const float hv = (float)Hmask[row * HH + n];
            outT[j][r] = (_Float16)((1.0f - hv * hv) * acc[r]);
        }
    }
#pragma unroll
    for (int j = 0; j < 16; ++j) {
        const int n = j * 16 + nlo;
#pragma unroll
        for (int r = 0; r < 8; ++r) {
            const int row = waveRow + rhi + r;
            dBuf[row * HH + n] = outT[j][r];
        }
    }
}

// Kernel 1: convert + transpose W2/W3 (f32 K-major) -> f16 N-major into d_ws.
__global__ void mhd_packW(const float* __restrict__ W2, const float* __restrict__ W3,
                          _Float16* __restrict__ wt2, _Float16* __restrict__ wt3) {
    int i = blockIdx.x * 256 + threadIdx.x;   // 65536 elements
    int n = i >> 8, k = i & 255;
    wt2[n * 256 + k] = (_Float16)W2[k * 256 + n];
    wt3[n * 256 + k] = (_Float16)W3[k * 256 + n];
}

// Kernel 2: fused MLP + 3 JVPs + MHD residuals, per 64-row block (4 waves).
__global__ __launch_bounds__(NTHR) void mhd_main(
    const float* __restrict__ coords,
    const float* __restrict__ W1, const float* __restrict__ b1,
    const float* __restrict__ b2, const float* __restrict__ b3,
    const float* __restrict__ W4, const float* __restrict__ b4,
    const _Float16* __restrict__ wt2, const _Float16* __restrict__ wt3,
    float* __restrict__ partials)
{
    extern __shared__ char smemRaw[];
    _Float16* sm = (_Float16*)smemRaw;
    _Float16* Wt = sm;                       // 65536
    _Float16* hA = sm + 65536;               // 16384
    _Float16* hB = hA + 16384;               // 16384
    _Float16* dD = hB + 16384;               // 3 x 16384 (in-place tangents)
    float* fArea = (float*)(sm + HALFS_TOTAL);
    float* w4s = fArea;                      // 1536
    float* b4s = w4s + 1536;                 // 6
    float* red = b4s + 6;                    // 7

    const int tid = threadIdx.x;
    const int lane = tid & 31;
    const int wave = tid >> 5;
    const int waveRow = wave * 16;
    const int rowBlock = blockIdx.x * ROWS;

    // Kick off async staging of Wt2 into LDS (overlaps with layer-1 math below).
    {
        const char* src = (const char*)wt2;
        char* dst = (char*)Wt;
        for (int i = tid; i < 8192; i += NTHR)
            async_copy_b128(src + i * 16, dst + i * 16);
    }

    if (tid < 7) red[tid] = 0.0f;
    // stage W4 / b4
    for (int i = tid; i < 1536; i += NTHR) w4s[i] = W4[i];
    if (tid < 6) b4s[tid] = b4[tid];

    // ---- layer 1: forward + 3 constant-direction tangents (scalar) ----
    for (int i = tid; i < ROWS * HH; i += NTHR) {
        const int row = i >> 8, col = i & 255;
        const float c0 = coords[(rowBlock + row) * 3 + 0];
        const float c1 = coords[(rowBlock + row) * 3 + 1];
        const float c2 = coords[(rowBlock + row) * 3 + 2];
        const float w0 = W1[0 * HH + col];
        const float w1 = W1[1 * HH + col];
        const float w2 = W1[2 * HH + col];
        const float pre = fmaf(c0, w0, fmaf(c1, w1, fmaf(c2, w2, b1[col])));
        const float h = my_tanhf(pre);
        const float m = 1.0f - h * h;
        hA[row * HH + col] = (_Float16)h;
        dD[0 * 16384 + row * HH + col] = (_Float16)(m * w0);  // d/dx
        dD[1 * 16384 + row * HH + col] = (_Float16)(m * w1);  // d/dy
        dD[2 * 16384 + row * HH + col] = (_Float16)(m * w2);  // d/dt
    }
    wait_async0();
    __syncthreads();

    _Float16* dA[3] = { dD, dD + 16384, dD + 2 * 16384 };

    // ---- layer 2 (WMMA): hA -> hB, dA[s] in place (mask = hB) ----
    forward_stream(hA, Wt, hB, b2, lane, waveRow);
    tangent_stream_inplace(dA[0], Wt, hB, lane, waveRow);
    tangent_stream_inplace(dA[1], Wt, hB, lane, waveRow);
    tangent_stream_inplace(dA[2], Wt, hB, lane, waveRow);
    __syncthreads();
    // async stage Wt3 over the now-dead Wt2
    {
        const char* src = (const char*)wt3;
        char* dst = (char*)Wt;
        for (int i = tid; i < 8192; i += NTHR)
            async_copy_b128(src + i * 16, dst + i * 16);
    }
    wait_async0();
    __syncthreads();
    // ---- layer 3 (WMMA): hB -> hA, dA[s] in place (mask = hA) ----
    forward_stream(hB, Wt, hA, b3, lane, waveRow);
    tangent_stream_inplace(dA[0], Wt, hA, lane, waveRow);
    tangent_stream_inplace(dA[1], Wt, hA, lane, waveRow);
    tangent_stream_inplace(dA[2], Wt, hA, lane, waveRow);
    __syncthreads();

    // ---- layer 4 (256x6 GEMV per row) + MHD residuals, one row per thread ----
    if (tid < ROWS) {
        const int row = tid;
        float pr[6], jx[6], jy[6], jt[6];
#pragma unroll
        for (int p = 0; p < 6; ++p) { pr[p] = b4s[p]; jx[p] = 0.f; jy[p] = 0.f; jt[p] = 0.f; }
        const _Float16* hrow = hA + row * HH;
        const _Float16* dxr = dA[0] + row * HH;
        const _Float16* dyr = dA[1] + row * HH;
        const _Float16* dtr = dA[2] + row * HH;
        for (int c = 0; c < HH; ++c) {
            const float hv = (float)hrow[c];
            const float xv = (float)dxr[c];
            const float yv = (float)dyr[c];
            const float tv = (float)dtr[c];
            const float* wrow = w4s + c * 6;
#pragma unroll
            for (int p = 0; p < 6; ++p) {
                const float w = wrow[p];
                pr[p] = fmaf(hv, w, pr[p]);
                jx[p] = fmaf(xv, w, jx[p]);
                jy[p] = fmaf(yv, w, jy[p]);
                jt[p] = fmaf(tv, w, jt[p]);
            }
        }
        const float rho = pr[0], vx = pr[1], vy = pr[2], Bx = pr[3], By = pr[4], P = pr[5];
        const float ig1 = 1.0f / (GAMMA - 1.0f);
        const float dt_rho = jt[0], dt_Bx = jt[3], dt_By = jt[4];
        const float dt_rhovx = dt_rho * vx + rho * jt[1];
        const float dt_rhovy = dt_rho * vy + rho * jt[2];
        const float v2 = vx * vx + vy * vy;
        auto dE = [&](const float* J) {
            return J[5] * ig1 + 0.5f * J[0] * v2 + rho * (vx * J[1] + vy * J[2])
                 + Bx * J[3] + By * J[4];
        };
        const float dE_dx = dE(jx), dE_dy = dE(jy), dE_dt = dE(jt);
        const float div_v = jx[1] + jy[2];
        const float div_B = jx[3] + jy[4];
        const float continuity = dt_rho + rho * div_v;
        const float dPm_dx = jx[5] + Bx * jx[3] + By * jx[4];
        const float dPm_dy = jy[5] + Bx * jy[3] + By * jy[4];
        const float momentum_x = dt_rhovx + dPm_dx - (Bx * jx[3] + By * jy[3]);
        const float momentum_y = dt_rhovy + dPm_dy - (Bx * jx[4] + By * jy[4]);
        auto dG = [&](const float* J) {
            return J[1] * By + vx * J[4] - J[2] * Bx - vy * J[3];
        };
        const float induction_x = dt_Bx + dG(jy);
        const float induction_y = dt_By - dG(jx);
        const float B2h = 0.5f * (Bx * Bx + By * By);
        const float E = P * ig1 + 0.5f * rho * v2 + B2h;
        const float S = E + P + B2h;
        const float dS_dx = dE_dx + jx[5] + Bx * jx[3] + By * jx[4];
        const float dS_dy = dE_dy + jy[5] + Bx * jy[3] + By * jy[4];
        const float D = Bx * vx + By * vy;
        auto dD_ = [&](const float* J) {
            return J[3] * vx + Bx * J[1] + J[4] * vy + By * J[2];
        };
        const float dFx_dx = dS_dx * vx + S * jx[1] - dD_(jx) * Bx - D * jx[3];
        const float dFy_dy = dS_dy * vy + S * jy[2] - dD_(jy) * By - D * jy[4];
        const float energy = dE_dt + dFx_dx + dFy_dy;

        const float res[7] = { continuity, momentum_x, momentum_y,
                               induction_x, induction_y, energy, div_B };
#pragma unroll
        for (int k = 0; k < 7; ++k) atomicAdd(&red[k], res[k] * res[k]);
    }
    __syncthreads();
    if (tid < 7) partials[blockIdx.x * 7 + tid] = red[tid];
}

// Kernel 3: reduce per-block partials -> scalar weighted loss.
__global__ void mhd_reduce(const float* __restrict__ partials,
                           const float* __restrict__ wts,
                           float* __restrict__ out)
{
    __shared__ float s[7 * 256];
    float a[7] = {0.f, 0.f, 0.f, 0.f, 0.f, 0.f, 0.f};
    for (int b = threadIdx.x; b < NBLK; b += 256)
#pragma unroll
        for (int k = 0; k < 7; ++k) a[k] += partials[b * 7 + k];
#pragma unroll
    for (int k = 0; k < 7; ++k) s[k * 256 + threadIdx.x] = a[k];
    __syncthreads();
    if (threadIdx.x == 0) {
        float loss = 0.f;
        const float invN = 1.0f / (float)NPTS;
        for (int k = 0; k < 7; ++k) {
            float t = 0.f;
            for (int i = 0; i < 256; ++i) t += s[k * 256 + i];
            loss += wts[k] * t * invN;
        }
        out[0] = loss;
    }
}

extern "C" void kernel_launch(void* const* d_in, const int* in_sizes, int n_in,
                              void* d_out, int out_size, void* d_ws, size_t ws_size,
                              hipStream_t stream) {
    const float* coords = (const float*)d_in[0];
    const float* W1 = (const float*)d_in[1];
    const float* b1 = (const float*)d_in[2];
    const float* W2 = (const float*)d_in[3];
    const float* b2 = (const float*)d_in[4];
    const float* W3 = (const float*)d_in[5];
    const float* b3 = (const float*)d_in[6];
    const float* W4 = (const float*)d_in[7];
    const float* b4 = (const float*)d_in[8];
    const float* wts = (const float*)d_in[9];
    float* out = (float*)d_out;

    _Float16* wt2 = (_Float16*)d_ws;
    _Float16* wt3 = wt2 + 65536;
    float* partials = (float*)((char*)d_ws + 2 * 65536 * sizeof(_Float16));

    mhd_packW<<<256, 256, 0, stream>>>(W2, W3, wt2, wt3);
    mhd_main<<<NBLK, NTHR, SMEM_BYTES, stream>>>(coords, W1, b1, b2, b3, W4, b4,
                                                 wt2, wt3, partials);
    mhd_reduce<<<1, 256, 0, stream>>>(partials, wts, out);
}
